// ConditionalCrossAttention_31696858644849
// MI455X (gfx1250) — matrix operand order repaired and checked by reading, not compiled, not run on hardware
//
#include <hip/hip_runtime.h>
#include <hip/hip_bf16.h>
#include <math.h>

// ---------- types ----------
typedef __attribute__((ext_vector_type(4)))  unsigned v4u;
typedef __attribute__((ext_vector_type(8)))  unsigned v8u;
typedef __attribute__((ext_vector_type(8)))  float    v8f;
typedef __attribute__((ext_vector_type(16))) __bf16   v16bf;

#define C_DIM 256
#define NHEAD 8
#define DH    32
#define NQ    300
#define NPAD  304   // 19 tiles of 16
#define LKV   4096
#define BATCH 16

#if defined(__has_builtin)
#if __has_builtin(__builtin_amdgcn_cvt_pk_bf16_f32)
#define HAVE_CVT_PK_BF16 1
#endif
#endif

// ---------- helpers ----------
__device__ __forceinline__ unsigned pk2(float lo, float hi) {
#ifdef HAVE_CVT_PK_BF16
  typedef __attribute__((ext_vector_type(2))) __bf16 v2bf;
  v2bf r = __builtin_amdgcn_cvt_pk_bf16_f32(lo, hi);
  return __builtin_bit_cast(unsigned, r);
#else
  unsigned a = __float_as_uint(lo);
  a = (a + 0x7FFFu + ((a >> 16) & 1u)) >> 16;
  unsigned b = __float_as_uint(hi);
  b = (b + 0x7FFFu + ((b >> 16) & 1u)) >> 16;
  return a | (b << 16);
#endif
}
__device__ __forceinline__ unsigned short bf16s(float f) {
  return (unsigned short)(pk2(f, f) & 0xffffu);
}
__device__ __forceinline__ v8f zero8() {
  v8f z;
#pragma unroll
  for (int i = 0; i < 8; ++i) z[i] = 0.f;
  return z;
}
__device__ __forceinline__ v8f wmma_bf16(v16bf a, v8u bu, v8f c) {
  return __builtin_amdgcn_wmma_f32_16x16x32_bf16(false, a, false,
                                                 __builtin_bit_cast(v16bf, bu),
                                                 (short)0, c, false, false);
}
__device__ __forceinline__ v16bf a_from_chunks(v4u lo, v4u hi) {
  v8u au = __builtin_shufflevector(lo, hi, 0, 1, 2, 3, 4, 5, 6, 7);
  return __builtin_bit_cast(v16bf, au);
}

// ---------- kernel: pack one 256x256 weight (y = x W^T => B[k][n] = W[n][k]) ----------
// Layout: Wp[((kt*16+nt)*32 + lane)*8 + j]  (bf16x2 words) -> per-lane b128 loads
__global__ __launch_bounds__(32) void k_pack_w(const float* __restrict__ W,
                                               unsigned* __restrict__ Wp) {
  const int tile = blockIdx.x;            // kt*16 + nt, 128 tiles
  const int lane = threadIdx.x;
  const int half = lane >> 4, ln = lane & 15;
  const int kt = tile >> 4, nt = tile & 15;
  const int n = nt * 16 + ln;
#pragma unroll
  for (int j = 0; j < 8; ++j) {
    const int k = kt * 32 + half * 16 + j * 2;   // B frag: lanes0-15 K0..15, 16-31 K16..31
    Wp[((size_t)tile * 32 + lane) * 8 + j] = pk2(W[n * 256 + k], W[n * 256 + k + 1]);
  }
}

// ---------- GEMM strip (8 col-tiles): batch-load 8 B frags, then 8 WMMAs ----------
__device__ __forceinline__ void gemm_strip8_f32(const float* __restrict__ Xrow,
                                                const unsigned* __restrict__ Wp,
                                                int lane, int ntBase, v8f acc[8]) {
  const int half = lane >> 4;
#pragma unroll
  for (int kt = 0; kt < 8; ++kt) {
    // A fragment: 16 contiguous floats (two 32B chunks), hw-packed to bf16
    const v8f x0 = *(const v8f*)(Xrow + kt * 32 + half * 16);
    const v8f x1 = *(const v8f*)(Xrow + kt * 32 + half * 16 + 8);
    v8u au;
#pragma unroll
    for (int j = 0; j < 4; ++j) {
      au[j]     = pk2(x0[2 * j], x0[2 * j + 1]);
      au[j + 4] = pk2(x1[2 * j], x1[2 * j + 1]);
    }
    const v16bf a = __builtin_bit_cast(v16bf, au);
    // 8 B fragments in distinct registers: one load clause, one wait, 8 WMMAs
    const unsigned* wb = Wp + ((size_t)(kt * 16 + ntBase) * 32 + lane) * 8;
    v8u bu[8];
#pragma unroll
    for (int nt = 0; nt < 8; ++nt) bu[nt] = *(const v8u*)(wb + (size_t)nt * 32 * 8);
#pragma unroll
    for (int nt = 0; nt < 8; ++nt) acc[nt] = wmma_bf16(a, bu[nt], acc[nt]);
  }
}

__device__ __forceinline__ void gemm_strip8_bf16(const unsigned short* __restrict__ Xrow,
                                                 const unsigned* __restrict__ Wp,
                                                 int lane, int ntBase, v8f acc[8]) {
  const int half = lane >> 4;
#pragma unroll
  for (int kt = 0; kt < 8; ++kt) {
    const v4u alo = *(const v4u*)(Xrow + kt * 32 + half * 16);
    const v4u ahi = *(const v4u*)(Xrow + kt * 32 + half * 16 + 8);
    const v16bf a = a_from_chunks(alo, ahi);
    const unsigned* wb = Wp + ((size_t)(kt * 16 + ntBase) * 32 + lane) * 8;
    v8u bu[8];
#pragma unroll
    for (int nt = 0; nt < 8; ++nt) bu[nt] = *(const v8u*)(wb + (size_t)nt * 32 * 8);
#pragma unroll
    for (int nt = 0; nt < 8; ++nt) acc[nt] = wmma_bf16(a, bu[nt], acc[nt]);
  }
}

// ---------- K-side: kc, kp, v projections -> K_cat[b,h,l,64], V_t[b,h,d,L] ----------
__global__ __launch_bounds__(128) void k_kside(
    const float* __restrict__ key, const float* __restrict__ key_pos,
    const float* __restrict__ value,
    const unsigned* __restrict__ Wkc, const unsigned* __restrict__ Wkp,
    const unsigned* __restrict__ Wv,
    const float* __restrict__ bkc, const float* __restrict__ bkp,
    const float* __restrict__ bv,
    unsigned short* __restrict__ Kcat, unsigned short* __restrict__ Vt) {
  __shared__ float lds[4][16][256];   // per-wave kp staging (64 KB)
  const int w = threadIdx.x >> 5, lane = threadIdx.x & 31;
  const int half = lane >> 4, ln = lane & 15;
  const int t0 = blockIdx.x * 64 + w * 16;
  const float* xkp = key_pos + (size_t)(t0 + ln) * C_DIM;
  const float* xkc = key + (size_t)(t0 + ln) * C_DIM;
  const float* xv  = value + (size_t)(t0 + ln) * C_DIM;

  // key_pos projection -> LDS
#pragma clang loop unroll(disable)
  for (int pass = 0; pass < 2; ++pass) {
    v8f acc[8];
#pragma unroll
    for (int nt = 0; nt < 8; ++nt) acc[nt] = zero8();
    gemm_strip8_f32(xkp, Wkp, lane, pass * 8, acc);
#pragma unroll
    for (int nt = 0; nt < 8; ++nt) {
      const int d = (pass * 8 + nt) * 16 + ln;
      const float bb = bkp[d];
#pragma unroll
      for (int v = 0; v < 8; ++v) lds[w][v + 8 * half][d] = acc[nt][v] + bb;
    }
  }

  // key content projection; K_cat = [kc+kp | kp]
#pragma clang loop unroll(disable)
  for (int pass = 0; pass < 2; ++pass) {
    v8f acc[8];
#pragma unroll
    for (int nt = 0; nt < 8; ++nt) acc[nt] = zero8();
    gemm_strip8_f32(xkc, Wkc, lane, pass * 8, acc);
#pragma unroll
    for (int nt = 0; nt < 8; ++nt) {
      const int d = (pass * 8 + nt) * 16 + ln;
      const int h = d >> 5, j = d & 31;
      const float bb = bkc[d];
#pragma unroll
      for (int v = 0; v < 8; ++v) {
        const int m = v + 8 * half;
        const int t = t0 + m, l = t >> 4, b = t & 15;
        const float kpv = lds[w][m][d];
        const size_t base = ((size_t)(b * NHEAD + h) * LKV + l) * 64;
        Kcat[base + j]      = bf16s(acc[nt][v] + bb + kpv);
        Kcat[base + 32 + j] = bf16s(kpv);
      }
    }
  }

  // value projection -> transposed V_t[b,h,d,L]
#pragma clang loop unroll(disable)
  for (int pass = 0; pass < 2; ++pass) {
    v8f acc[8];
#pragma unroll
    for (int nt = 0; nt < 8; ++nt) acc[nt] = zero8();
    gemm_strip8_f32(xv, Wv, lane, pass * 8, acc);
#pragma unroll
    for (int nt = 0; nt < 8; ++nt) {
      const int d = (pass * 8 + nt) * 16 + ln;
      const int h = d >> 5, j = d & 31;
      const float bb = bv[d];
#pragma unroll
      for (int v = 0; v < 8; ++v) {
        const int m = v + 8 * half;
        const int t = t0 + m, l = t >> 4, b = t & 15;
        Vt[((size_t)(b * NHEAD + h) * DH + j) * LKV + l] = bf16s(acc[nt][v] + bb);
      }
    }
  }
}

// ---------- Q-side: qc, qp, qs -> Q_cat[b,h,n,64] (pre-scaled by 1/8) ----------
__global__ __launch_bounds__(128) void k_qside(
    const float* __restrict__ query, const float* __restrict__ query_pos,
    const float* __restrict__ qsine,
    const unsigned* __restrict__ Wqc, const unsigned* __restrict__ Wqp,
    const unsigned* __restrict__ Wqs,
    const float* __restrict__ bqc, const float* __restrict__ bqp,
    const float* __restrict__ bqs,
    unsigned short* __restrict__ Qcat) {
  __shared__ float lds[4][16][256];
  const int w = threadIdx.x >> 5, lane = threadIdx.x & 31;
  const int half = lane >> 4, ln = lane & 15;
  const int t0 = blockIdx.x * 64 + w * 16;   // 75 blocks * 64 = 4800 exactly
  const float* xqp = query_pos + (size_t)(t0 + ln) * C_DIM;
  const float* xqc = query + (size_t)(t0 + ln) * C_DIM;
  const float* xqs = qsine + (size_t)(t0 + ln) * C_DIM;

#pragma clang loop unroll(disable)
  for (int pass = 0; pass < 2; ++pass) {
    v8f acc[8];
#pragma unroll
    for (int nt = 0; nt < 8; ++nt) acc[nt] = zero8();
    gemm_strip8_f32(xqp, Wqp, lane, pass * 8, acc);
#pragma unroll
    for (int nt = 0; nt < 8; ++nt) {
      const int d = (pass * 8 + nt) * 16 + ln;
      const float bb = bqp[d];
#pragma unroll
      for (int v = 0; v < 8; ++v) lds[w][v + 8 * half][d] = acc[nt][v] + bb;
    }
  }

#pragma clang loop unroll(disable)
  for (int pass = 0; pass < 2; ++pass) {
    v8f acc[8];
#pragma unroll
    for (int nt = 0; nt < 8; ++nt) acc[nt] = zero8();
    gemm_strip8_f32(xqc, Wqc, lane, pass * 8, acc);
#pragma unroll
    for (int nt = 0; nt < 8; ++nt) {
      const int d = (pass * 8 + nt) * 16 + ln;
      const int h = d >> 5, j = d & 31;
      const float bb = bqc[d];
#pragma unroll
      for (int v = 0; v < 8; ++v) {
        const int m = v + 8 * half;
        const int t = t0 + m, n = t >> 4, b = t & 15;
        const float q1 = (acc[nt][v] + bb + lds[w][m][d]) * 0.125f;
        Qcat[((size_t)(b * NHEAD + h) * NPAD + n) * 64 + j] = bf16s(q1);
      }
    }
  }

#pragma clang loop unroll(disable)
  for (int pass = 0; pass < 2; ++pass) {
    v8f acc[8];
#pragma unroll
    for (int nt = 0; nt < 8; ++nt) acc[nt] = zero8();
    gemm_strip8_f32(xqs, Wqs, lane, pass * 8, acc);
#pragma unroll
    for (int nt = 0; nt < 8; ++nt) {
      const int d = (pass * 8 + nt) * 16 + ln;
      const int h = d >> 5, j = d & 31;
      const float bb = bqs[d];
#pragma unroll
      for (int v = 0; v < 8; ++v) {
        const int m = v + 8 * half;
        const int t = t0 + m, n = t >> 4, b = t & 15;
        Qcat[((size_t)(b * NHEAD + h) * NPAD + n) * 64 + 32 + j] =
            bf16s((acc[nt][v] + bb) * 0.125f);
      }
    }
  }
}

// ---------- flash attention: grid (19 q-tiles, 128 bh), 4 waves split L ----------
__global__ __launch_bounds__(128) void k_attn(
    const unsigned short* __restrict__ Qcat, const unsigned short* __restrict__ Kcat,
    const unsigned short* __restrict__ Vt, unsigned short* __restrict__ Aout) {
  __shared__ unsigned short pb[4][16][32];   // per-wave P transpose buffer
  __shared__ float Om[4][16][32];            // per-wave O partials
  __shared__ float Mx[4][16];
  __shared__ float Sm[4][16];
  const int qt = blockIdx.x, bh = blockIdx.y;
  const int w = threadIdx.x >> 5, lane = threadIdx.x & 31;
  const int half = lane >> 4, ln = lane & 15;

  // Q A-fragments (d 0..31 and 32..63), loaded once (contiguous 16B chunks)
  const unsigned short* qrow = Qcat + ((size_t)bh * NPAD + qt * 16 + ln) * 64;
  const v16bf aq0 = a_from_chunks(*(const v4u*)(qrow + half * 16),
                                  *(const v4u*)(qrow + half * 16 + 8));
  const v16bf aq1 = a_from_chunks(*(const v4u*)(qrow + 32 + half * 16),
                                  *(const v4u*)(qrow + 32 + half * 16 + 8));

  float rm[8], rs[8];
#pragma unroll
  for (int v = 0; v < 8; ++v) { rm[v] = -3.0e38f; rs[v] = 0.f; }
  v8f vo0 = zero8(), vo1 = zero8();

  const unsigned short* Kb = Kcat + (size_t)bh * LKV * 64;
  const unsigned short* Vb = Vt + (size_t)bh * DH * LKV + half * 16;

  const int lend = w * 1024 + 1024;
  for (int l0 = w * 1024; l0 < lend; l0 += 32) {
    // ---- issue all K and V fragment loads up front (each a 32B chunk) ----
    const unsigned short* kr0 = Kb + (size_t)(l0 + ln) * 64 + half * 16;
    const unsigned short* kr1 = Kb + (size_t)(l0 + 16 + ln) * 64 + half * 16;
    const v8u kb00 = *(const v8u*)(kr0);
    const v8u kb01 = *(const v8u*)(kr0 + 32);
    const v8u kb10 = *(const v8u*)(kr1);
    const v8u kb11 = *(const v8u*)(kr1 + 32);
    const v8u vb0 = *(const v8u*)(Vb + (size_t)ln * LKV + l0);
    const v8u vb1 = *(const v8u*)(Vb + (size_t)(16 + ln) * LKV + l0);
    // ---- scores: two 16x16 tiles over K-dim 64 ----
    v8f s0 = wmma_bf16(aq0, kb00, zero8());
    s0 = wmma_bf16(aq1, kb01, s0);
    v8f s1 = wmma_bf16(aq0, kb10, zero8());
    s1 = wmma_bf16(aq1, kb11, s1);
    // ---- online softmax (row = v + 8*half, cols across 16 lanes of half) ----
#pragma unroll
    for (int v = 0; v < 8; ++v) {
      const float e0 = s0[v], e1 = s1[v];
      float mx = fmaxf(e0, e1);
      mx = fmaxf(mx, __shfl_xor(mx, 1, 32));
      mx = fmaxf(mx, __shfl_xor(mx, 2, 32));
      mx = fmaxf(mx, __shfl_xor(mx, 4, 32));
      mx = fmaxf(mx, __shfl_xor(mx, 8, 32));
      const float nm = fmaxf(rm[v], mx);
      const float corr = __expf(rm[v] - nm);
      rm[v] = nm;
      const float p0 = __expf(e0 - nm), p1 = __expf(e1 - nm);
      float ps = p0 + p1;
      ps += __shfl_xor(ps, 1, 32);
      ps += __shfl_xor(ps, 2, 32);
      ps += __shfl_xor(ps, 4, 32);
      ps += __shfl_xor(ps, 8, 32);
      rs[v] = rs[v] * corr + ps;
      vo0[v] *= corr; vo1[v] *= corr;
      const int m = v + 8 * half;
      pb[w][m][ln]      = bf16s(p0);
      pb[w][m][16 + ln] = bf16s(p1);
    }
    asm volatile("s_wait_dscnt 0" ::: "memory");   // cross-lane LDS transpose hop
    // ---- P as A-fragment (16x32), PV WMMAs (V already resident) ----
    const v16bf ap = a_from_chunks(*(const v4u*)(&pb[w][ln][half * 16]),
                                   *(const v4u*)(&pb[w][ln][half * 16 + 8]));
    vo0 = wmma_bf16(ap, vb0, vo0);
    vo1 = wmma_bf16(ap, vb1, vo1);
  }

  // ---- merge 4 wave partials (log-sum-exp) ----
#pragma unroll
  for (int v = 0; v < 8; ++v) {
    const int m = v + 8 * half;
    Om[w][m][ln] = vo0[v];
    Om[w][m][16 + ln] = vo1[v];
  }
  if (ln == 0) {
#pragma unroll
    for (int v = 0; v < 8; ++v) { Mx[w][v + 8 * half] = rm[v]; Sm[w][v + 8 * half] = rs[v]; }
  }
  __syncthreads();
  const int b = bh >> 3, h = bh & 7;
  for (int idx = threadIdx.x; idx < 512; idx += 128) {
    const int q = idx >> 5, d = idx & 31;
    float M = fmaxf(fmaxf(Mx[0][q], Mx[1][q]), fmaxf(Mx[2][q], Mx[3][q]));
    float den = 0.f, num = 0.f;
#pragma unroll
    for (int ww = 0; ww < 4; ++ww) {
      const float e = __expf(Mx[ww][q] - M);
      den += Sm[ww][q] * e;
      num += Om[ww][q][d] * e;
    }
    const int nq = qt * 16 + q;
    if (nq < NQ)
      Aout[((size_t)b * NQ + nq) * C_DIM + h * DH + d] = bf16s(num / den);
  }
}

// ---------- output projection + bias + residual ----------
__global__ __launch_bounds__(128) void k_oproj(
    const unsigned short* __restrict__ Aout, const unsigned* __restrict__ Wo,
    const float* __restrict__ bo, const float* __restrict__ query,
    float* __restrict__ out) {
  const int w = threadIdx.x >> 5, lane = threadIdx.x & 31;
  const int half = lane >> 4, ln = lane & 15;
  const int t0 = blockIdx.x * 64 + w * 16;
  const int tr = t0 + ln;                                     // this lane's A row
  const unsigned short* Xrow = Aout + ((size_t)(tr & 15) * NQ + (tr >> 4)) * C_DIM;

#pragma clang loop unroll(disable)
  for (int pass = 0; pass < 2; ++pass) {
    v8f acc[8];
#pragma unroll
    for (int nt = 0; nt < 8; ++nt) acc[nt] = zero8();
    gemm_strip8_bf16(Xrow, Wo, lane, pass * 8, acc);
#pragma unroll
    for (int nt = 0; nt < 8; ++nt) {
      const int d = (pass * 8 + nt) * 16 + ln;
      const float bb = bo[d];
#pragma unroll
      for (int v = 0; v < 8; ++v) {
        const int t = t0 + v + 8 * half;                      // [N,B,C]: n*16+b == t
        out[(size_t)t * C_DIM + d] = acc[nt][v] + bb + query[(size_t)t * C_DIM + d];
      }
    }
  }
}

// ---------- host ----------
extern "C" void kernel_launch(void* const* d_in, const int* in_sizes, int n_in,
                              void* d_out, int out_size, void* d_ws, size_t ws_size,
                              hipStream_t stream) {
  (void)in_sizes; (void)n_in; (void)out_size; (void)ws_size;
  char* ws = (char*)d_ws;

  const size_t WB = 7u * 131072u;                                   // packed weights
  const size_t QB = (size_t)BATCH * NHEAD * NPAD * 64 * 2;          // Q_cat bf16
  const size_t KB = (size_t)BATCH * NHEAD * LKV * 64 * 2;           // K_cat bf16
  const size_t VB = (size_t)BATCH * NHEAD * DH * LKV * 2;           // V_t bf16

  unsigned* Wp = (unsigned*)ws;
  unsigned short* Qcat = (unsigned short*)(ws + WB);
  unsigned short* Kcat = (unsigned short*)(ws + WB + QB);
  unsigned short* Vt   = (unsigned short*)(ws + WB + QB + KB);
  unsigned short* Aout = (unsigned short*)(ws + WB + QB + KB + VB);

  // pack weights: order qc, qp, qs, kc, kp, v, o
  const int widx[7] = {6, 8, 10, 12, 14, 16, 18};
  for (int m = 0; m < 7; ++m)
    k_pack_w<<<128, 32, 0, stream>>>((const float*)d_in[widx[m]],
                                     Wp + (size_t)m * 32768);

  hipMemsetAsync(Qcat, 0, QB, stream);   // zero padded query rows 300..303

  k_kside<<<1024, 128, 0, stream>>>(
      (const float*)d_in[1], (const float*)d_in[4], (const float*)d_in[2],
      Wp + 3 * 32768, Wp + 4 * 32768, Wp + 5 * 32768,
      (const float*)d_in[13], (const float*)d_in[15], (const float*)d_in[17],
      Kcat, Vt);

  k_qside<<<75, 128, 0, stream>>>(
      (const float*)d_in[0], (const float*)d_in[3], (const float*)d_in[5],
      Wp + 0 * 32768, Wp + 1 * 32768, Wp + 2 * 32768,
      (const float*)d_in[7], (const float*)d_in[9], (const float*)d_in[11],
      Qcat);

  k_attn<<<dim3(19, 128), 128, 0, stream>>>(Qcat, Kcat, Vt, Aout);

  k_oproj<<<75, 128, 0, stream>>>(Aout, Wp + 6 * 32768,
                                  (const float*)d_in[19], (const float*)d_in[0],
                                  (float*)d_out);
}
